// LinearBlock_19284403159676
// MI455X (gfx1250) — matrix operand order repaired and checked by reading, not compiled, not run on hardware
//
#include <hip/hip_runtime.h>

// ---------------------------------------------------------------------------
// BN(train) + Linear + LeakyReLU fused pipeline for gfx1250 (MI455X).
//
// y = leaky( (x * s) . W^T + c ),  s = gamma*rsqrt(var+eps),
//                                  c = b + (beta - mean*s) . W^T
//
// GEMM runs on v_wmma_f32_16x16x32_bf16 with a bf16 hi/lo error-compensated
// split (hi*hi + hi*lo + lo*hi) for near-fp32 accuracy at matrix-core speed.
//
// Preferred path (if workspace is large enough): pre-convert x*s and W once
// into four bf16 planes, then the GEMM inner loop is async global->LDS
// copies (ASYNCcnt) double-buffered against WMMA — near-zero VALU.
// Fallback path: convert inline in the GEMM (round-1 kernel).
// ---------------------------------------------------------------------------

typedef __attribute__((ext_vector_type(16))) __bf16 v16bf;
typedef __attribute__((ext_vector_type(8)))  float  v8f;
typedef __attribute__((ext_vector_type(4)))  int    v4i;

// address-space-qualified pointee types for the async-copy builtin
typedef __attribute__((address_space(1))) v4i gv4i;   // global
typedef __attribute__((address_space(3))) v4i lv4i;   // LDS

constexpr int   BATCH  = 8192;
constexpr int   IN     = 4096;
constexpr int   OUTF   = 4096;
constexpr int   SPLITS = 32;      // row splits for deterministic BN reduction
constexpr float BN_EPS = 1e-5f;
constexpr float SLOPE  = 0.01f;

// GEMM tiling
constexpr int BM  = 128;   // rows per workgroup tile
constexpr int BN_ = 128;   // cols per workgroup tile
constexpr int BK  = 32;    // K per stage (one bf16 WMMA K)
constexpr int LDT = 40;    // padded LDS row stride (elements) -> 80B rows, 16B aligned

__device__ __forceinline__ unsigned short f2bf_rne(float f) {
  unsigned u = __float_as_uint(f);
  unsigned r = (u + 0x7FFFu + ((u >> 16) & 1u)) >> 16;   // round-to-nearest-even
  return (unsigned short)r;
}
__device__ __forceinline__ float bf2f(unsigned short h) {
  return __uint_as_float(((unsigned)h) << 16);
}

union FragBF { uint4 q[2]; v16bf v; };

// ---- CDNA5 async global->LDS copy (tracked by ASYNCcnt) -------------------
#if __has_builtin(__builtin_amdgcn_global_load_async_to_lds_b128)
#define HAVE_ASYNC_LDS 1
__device__ __forceinline__ void async_copy16(const void* g, void* l) {
  __builtin_amdgcn_global_load_async_to_lds_b128((gv4i*)(g), (lv4i*)(l), 0, 0);
}
#else
#define HAVE_ASYNC_LDS 0
__device__ __forceinline__ void async_copy16(const void* g, void* l) {
  *(uint4*)l = *(const uint4*)g;      // synchronous fallback
}
#endif

__device__ __forceinline__ void wait_async_copies() {
#if HAVE_ASYNC_LDS
#if __has_builtin(__builtin_amdgcn_s_wait_asynccnt)
  __builtin_amdgcn_s_wait_asynccnt(0);
#else
  asm volatile("s_wait_asynccnt 0" ::: "memory");
#endif
#endif
}

// ---------------------------------------------------------------------------
// Stage 1: per-column partial sum / sum-of-squares over 256-row slabs.
// ---------------------------------------------------------------------------
__global__ void bn_partial_kernel(const float* __restrict__ x,
                                  float* __restrict__ psum,
                                  float* __restrict__ psq) {
  const int col = blockIdx.x * blockDim.x + threadIdx.x;
  const int r0  = blockIdx.y * (BATCH / SPLITS);
  float s = 0.f, q = 0.f;
  for (int r = 0; r < BATCH / SPLITS; ++r) {
    float v = x[(size_t)(r0 + r) * IN + col];
    s += v;
    q += v * v;
  }
  psum[(size_t)blockIdx.y * IN + col] = s;
  psq [(size_t)blockIdx.y * IN + col] = q;
}

// ---------------------------------------------------------------------------
// Stage 2: deterministic reduction of SPLITS partials -> s[i], t[i]
// ---------------------------------------------------------------------------
__global__ void bn_finalize_kernel(const float* __restrict__ psum,
                                   const float* __restrict__ psq,
                                   const float* __restrict__ gamma,
                                   const float* __restrict__ beta,
                                   float* __restrict__ s_out,
                                   float* __restrict__ t_out) {
  const int col = blockIdx.x * blockDim.x + threadIdx.x;
  float s = 0.f, q = 0.f;
  for (int i = 0; i < SPLITS; ++i) {
    s += psum[(size_t)i * IN + col];
    q += psq [(size_t)i * IN + col];
  }
  const float inv_n = 1.0f / (float)BATCH;
  const float mean  = s * inv_n;
  const float var   = q * inv_n - mean * mean;
  const float rstd  = rsqrtf(var + BN_EPS);
  const float sc    = gamma[col] * rstd;
  s_out[col] = sc;
  t_out[col] = beta[col] - mean * sc;
}

// ---------------------------------------------------------------------------
// Stage 3: fold BN shift into per-output constant c[o] = b[o] + sum_i t[i]*W[o,i]
// ---------------------------------------------------------------------------
__global__ void fold_bias_kernel(const float* __restrict__ t,
                                 const float* __restrict__ W,
                                 const float* __restrict__ bias,
                                 float* __restrict__ cb) {
  __shared__ float red[256];
  const int o   = blockIdx.x;
  const int tid = threadIdx.x;
  float p = 0.f;
  const float* wr = W + (size_t)o * IN;
  for (int i = tid; i < IN; i += 256) p += t[i] * wr[i];
  red[tid] = p;
  __syncthreads();
  for (int ofs = 128; ofs > 0; ofs >>= 1) {
    if (tid < ofs) red[tid] += red[tid + ofs];
    __syncthreads();
  }
  if (tid == 0) cb[o] = bias[o] + red[0];
}

// ---------------------------------------------------------------------------
// Stage 3b (packed path): one-shot fp32 -> bf16 hi/lo plane conversion.
// ---------------------------------------------------------------------------
__global__ void pack_x_kernel(const float* __restrict__ x,
                              const float* __restrict__ s,
                              unsigned short* __restrict__ HI,
                              unsigned short* __restrict__ LO) {
  const size_t base = ((size_t)blockIdx.x * blockDim.x + threadIdx.x) * 4;
  const int col = (int)(base & (size_t)(IN - 1));
  float4 xv = *(const float4*)(x + base);
  float4 sv = *(const float4*)(s + col);
  xv.x *= sv.x; xv.y *= sv.y; xv.z *= sv.z; xv.w *= sv.w;
  const float vv[4] = {xv.x, xv.y, xv.z, xv.w};
  unsigned short h[4], l[4];
  #pragma unroll
  for (int j = 0; j < 4; ++j) {
    h[j] = f2bf_rne(vv[j]);
    l[j] = f2bf_rne(vv[j] - bf2f(h[j]));
  }
  *(uint2*)&HI[base] = make_uint2((unsigned)h[0] | ((unsigned)h[1] << 16),
                                  (unsigned)h[2] | ((unsigned)h[3] << 16));
  *(uint2*)&LO[base] = make_uint2((unsigned)l[0] | ((unsigned)l[1] << 16),
                                  (unsigned)l[2] | ((unsigned)l[3] << 16));
}

__global__ void pack_w_kernel(const float* __restrict__ W,
                              unsigned short* __restrict__ HI,
                              unsigned short* __restrict__ LO) {
  const size_t base = ((size_t)blockIdx.x * blockDim.x + threadIdx.x) * 4;
  float4 wv = *(const float4*)(W + base);
  const float vv[4] = {wv.x, wv.y, wv.z, wv.w};
  unsigned short h[4], l[4];
  #pragma unroll
  for (int j = 0; j < 4; ++j) {
    h[j] = f2bf_rne(vv[j]);
    l[j] = f2bf_rne(vv[j] - bf2f(h[j]));
  }
  *(uint2*)&HI[base] = make_uint2((unsigned)h[0] | ((unsigned)h[1] << 16),
                                  (unsigned)h[2] | ((unsigned)h[3] << 16));
  *(uint2*)&LO[base] = make_uint2((unsigned)l[0] | ((unsigned)l[1] << 16),
                                  (unsigned)l[2] | ((unsigned)l[3] << 16));
}

// ---------------------------------------------------------------------------
// Stage 4 (packed path): WMMA GEMM over pre-converted bf16 planes.
// 128x128 tile / 256 threads (8 wave32s, 2x4). Double-buffered LDS staged
// with async global->LDS b128 copies overlapped against WMMA compute.
// ---------------------------------------------------------------------------
__device__ __forceinline__ void stage_tile_async(const unsigned short* __restrict__ plane,
                                                 unsigned short* __restrict__ lbuf,
                                                 int row0, int k0, int tid) {
  // 128 rows x 32 bf16 = 512 x 16B chunks; 256 threads x 2 chunks
  #pragma unroll
  for (int j = 0; j < 2; ++j) {
    const int c   = tid + j * 256;
    const int row = c >> 2;
    const int ko  = (c & 3) * 8;
    async_copy16(plane + (size_t)(row0 + row) * IN + k0 + ko,
                 &lbuf[row * LDT + ko]);
  }
}

__global__ __launch_bounds__(256)
void gemm_packed_wmma_kernel(const unsigned short* __restrict__ HA,
                             const unsigned short* __restrict__ LA,
                             const unsigned short* __restrict__ HB,
                             const unsigned short* __restrict__ LB,
                             const float* __restrict__ cb,
                             float* __restrict__ out) {
  __shared__ __align__(16) unsigned short Ah[2][BM * LDT];
  __shared__ __align__(16) unsigned short Al[2][BM * LDT];
  __shared__ __align__(16) unsigned short Bh[2][BN_ * LDT];
  __shared__ __align__(16) unsigned short Bl[2][BN_ * LDT];

  const int tid  = threadIdx.x;
  const int lane = tid & 31;
  const int wid  = tid >> 5;
  const int wm   = wid >> 2;     // 0..1 : 64-row band
  const int wn   = wid & 3;      // 0..3 : 32-col band
  const int lr   = lane & 15;
  const int hf   = lane >> 4;

  const int m0 = blockIdx.y * BM;
  const int n0 = blockIdx.x * BN_;

  v8f acc[4][2] = {};

  // prologue: stage tile 0 into buffer 0
  stage_tile_async(HA, Ah[0], m0, 0, tid);
  stage_tile_async(LA, Al[0], m0, 0, tid);
  stage_tile_async(HB, Bh[0], n0, 0, tid);
  stage_tile_async(LB, Bl[0], n0, 0, tid);
  wait_async_copies();
  __syncthreads();

  constexpr int KT = IN / BK;
  for (int kt = 0; kt < KT; ++kt) {
    const int p = kt & 1;

    // kick off next tile's async copies into the other buffer
    if (kt + 1 < KT) {
      const int k1 = (kt + 1) * BK;
      stage_tile_async(HA, Ah[p ^ 1], m0, k1, tid);
      stage_tile_async(LA, Al[p ^ 1], m0, k1, tid);
      stage_tile_async(HB, Bh[p ^ 1], n0, k1, tid);
      stage_tile_async(LB, Bl[p ^ 1], n0, k1, tid);
    }

    // ---- compute on buffer p (A/B lane half hf holds k = 16*hf + s) -----
    FragBF bh[2], bl[2];
    #pragma unroll
    for (int nt = 0; nt < 2; ++nt) {
      const int brow = (wn * 32 + nt * 16 + lr) * LDT + hf * 16;
      const uint4* bp = (const uint4*)&Bh[p][brow];
      bh[nt].q[0] = bp[0]; bh[nt].q[1] = bp[1];
      const uint4* bq = (const uint4*)&Bl[p][brow];
      bl[nt].q[0] = bq[0]; bl[nt].q[1] = bq[1];
    }

    #pragma unroll
    for (int mt = 0; mt < 4; ++mt) {
      FragBF ah, al;
      const int ar = (wm * 64 + mt * 16 + lr) * LDT;
      ah.q[0] = *(const uint4*)&Ah[p][ar + hf * 16];
      ah.q[1] = *(const uint4*)&Ah[p][ar + hf * 16 + 8];
      al.q[0] = *(const uint4*)&Al[p][ar + hf * 16];
      al.q[1] = *(const uint4*)&Al[p][ar + hf * 16 + 8];

      #pragma unroll
      for (int nt = 0; nt < 2; ++nt) {
        acc[mt][nt] = __builtin_amdgcn_wmma_f32_16x16x32_bf16(
            false, ah.v, false, bh[nt].v, (short)0, acc[mt][nt], false, false);
        acc[mt][nt] = __builtin_amdgcn_wmma_f32_16x16x32_bf16(
            false, ah.v, false, bl[nt].v, (short)0, acc[mt][nt], false, false);
        acc[mt][nt] = __builtin_amdgcn_wmma_f32_16x16x32_bf16(
            false, al.v, false, bh[nt].v, (short)0, acc[mt][nt], false, false);
      }
    }

    wait_async_copies();   // next tile landed
    __syncthreads();       // all waves done reading buffer p / copies visible
  }

  // ---- epilogue: + folded constant, LeakyReLU, fp32 store -----------------
  #pragma unroll
  for (int mt = 0; mt < 4; ++mt) {
    #pragma unroll
    for (int nt = 0; nt < 2; ++nt) {
      const int gn = n0 + wn * 32 + nt * 16 + lr;
      const float c0 = cb[gn];
      #pragma unroll
      for (int r = 0; r < 8; ++r) {
        const int gm = m0 + wm * 64 + mt * 16 + hf * 8 + r;
        float y = acc[mt][nt][r] + c0;
        out[(size_t)gm * OUTF + gn] = (y >= 0.f) ? y : SLOPE * y;
      }
    }
  }
}

// ---------------------------------------------------------------------------
// Stage 4 (fallback path): inline-conversion WMMA GEMM (round-1 kernel).
// ---------------------------------------------------------------------------
__global__ __launch_bounds__(256)
void gemm_bn_wmma_kernel(const float* __restrict__ x,
                         const float* __restrict__ W,
                         const float* __restrict__ s,
                         const float* __restrict__ cb,
                         float* __restrict__ out) {
  __shared__ __align__(16) unsigned short Ahi[BM * LDT];
  __shared__ __align__(16) unsigned short Alo[BM * LDT];
  __shared__ __align__(16) unsigned short Bhi[BN_ * LDT];
  __shared__ __align__(16) unsigned short Blo[BN_ * LDT];

  const int tid  = threadIdx.x;
  const int lane = tid & 31;
  const int wid  = tid >> 5;
  const int wm   = wid >> 2;
  const int wn   = wid & 3;
  const int lr   = lane & 15;
  const int hf   = lane >> 4;

  const int m0 = blockIdx.y * BM;
  const int n0 = blockIdx.x * BN_;

  v8f acc[4][2] = {};

  for (int kt = 0; kt < IN / BK; ++kt) {
    const int k0 = kt * BK;

    #pragma unroll
    for (int i = 0; i < 4; ++i) {
      const int pos = tid + i * 256;
      const int row = pos >> 3;
      const int kq  = (pos & 7) * 4;
      float4 xv = *(const float4*)(x + (size_t)(m0 + row) * IN + k0 + kq);
      float4 sv = *(const float4*)(s + k0 + kq);
      xv.x *= sv.x; xv.y *= sv.y; xv.z *= sv.z; xv.w *= sv.w;
      const float vv[4] = {xv.x, xv.y, xv.z, xv.w};
      unsigned short h[4], l[4];
      #pragma unroll
      for (int j = 0; j < 4; ++j) {
        h[j] = f2bf_rne(vv[j]);
        l[j] = f2bf_rne(vv[j] - bf2f(h[j]));
      }
      *(uint2*)&Ahi[row * LDT + kq] =
          make_uint2((unsigned)h[0] | ((unsigned)h[1] << 16),
                     (unsigned)h[2] | ((unsigned)h[3] << 16));
      *(uint2*)&Alo[row * LDT + kq] =
          make_uint2((unsigned)l[0] | ((unsigned)l[1] << 16),
                     (unsigned)l[2] | ((unsigned)l[3] << 16));
    }

    #pragma unroll
    for (int i = 0; i < 4; ++i) {
      const int pos = tid + i * 256;
      const int row = pos >> 3;
      const int kq  = (pos & 7) * 4;
      float4 wv = *(const float4*)(W + (size_t)(n0 + row) * IN + k0 + kq);
      const float vv[4] = {wv.x, wv.y, wv.z, wv.w};
      unsigned short h[4], l[4];
      #pragma unroll
      for (int j = 0; j < 4; ++j) {
        h[j] = f2bf_rne(vv[j]);
        l[j] = f2bf_rne(vv[j] - bf2f(h[j]));
      }
      *(uint2*)&Bhi[row * LDT + kq] =
          make_uint2((unsigned)h[0] | ((unsigned)h[1] << 16),
                     (unsigned)h[2] | ((unsigned)h[3] << 16));
      *(uint2*)&Blo[row * LDT + kq] =
          make_uint2((unsigned)l[0] | ((unsigned)l[1] << 16),
                     (unsigned)l[2] | ((unsigned)l[3] << 16));
    }

    if (kt + 1 < IN / BK) {
      __builtin_prefetch(x + (size_t)(m0 + (tid >> 1)) * IN + k0 + BK, 0, 1);
      __builtin_prefetch(W + (size_t)(n0 + (tid >> 1)) * IN + k0 + BK, 0, 1);
    }

    __syncthreads();

    FragBF bh[2], bl[2];
    #pragma unroll
    for (int nt = 0; nt < 2; ++nt) {
      const int brow = (wn * 32 + nt * 16 + lr) * LDT + hf * 16;
      const uint4* bp = (const uint4*)&Bhi[brow];
      bh[nt].q[0] = bp[0]; bh[nt].q[1] = bp[1];
      const uint4* bq = (const uint4*)&Blo[brow];
      bl[nt].q[0] = bq[0]; bl[nt].q[1] = bq[1];
    }

    #pragma unroll
    for (int mt = 0; mt < 4; ++mt) {
      FragBF ah, al;
      const int ar = (wm * 64 + mt * 16 + lr) * LDT;
      ah.q[0] = *(const uint4*)&Ahi[ar + hf * 16];
      ah.q[1] = *(const uint4*)&Ahi[ar + hf * 16 + 8];
      al.q[0] = *(const uint4*)&Alo[ar + hf * 16];
      al.q[1] = *(const uint4*)&Alo[ar + hf * 16 + 8];

      #pragma unroll
      for (int nt = 0; nt < 2; ++nt) {
        acc[mt][nt] = __builtin_amdgcn_wmma_f32_16x16x32_bf16(
            false, ah.v, false, bh[nt].v, (short)0, acc[mt][nt], false, false);
        acc[mt][nt] = __builtin_amdgcn_wmma_f32_16x16x32_bf16(
            false, ah.v, false, bl[nt].v, (short)0, acc[mt][nt], false, false);
        acc[mt][nt] = __builtin_amdgcn_wmma_f32_16x16x32_bf16(
            false, al.v, false, bh[nt].v, (short)0, acc[mt][nt], false, false);
      }
    }

    __syncthreads();
  }

  #pragma unroll
  for (int mt = 0; mt < 4; ++mt) {
    #pragma unroll
    for (int nt = 0; nt < 2; ++nt) {
      const int gn = n0 + wn * 32 + nt * 16 + lr;
      const float c0 = cb[gn];
      #pragma unroll
      for (int r = 0; r < 8; ++r) {
        const int gm = m0 + wm * 64 + mt * 16 + hf * 8 + r;
        float y = acc[mt][nt][r] + c0;
        out[(size_t)gm * OUTF + gn] = (y >= 0.f) ? y : SLOPE * y;
      }
    }
  }
}

// ---------------------------------------------------------------------------
extern "C" void kernel_launch(void* const* d_in, const int* in_sizes, int n_in,
                              void* d_out, int out_size, void* d_ws, size_t ws_size,
                              hipStream_t stream) {
  const float* x     = (const float*)d_in[0];   // [8192, 4096]
  const float* gamma = (const float*)d_in[1];   // [4096]
  const float* beta  = (const float*)d_in[2];   // [4096]
  const float* W     = (const float*)d_in[3];   // [4096, 4096]
  const float* bias  = (const float*)d_in[4];   // [4096]
  float* out = (float*)d_out;                   // [8192, 4096]

  // workspace layout (floats): s[IN] | t[IN] | c[OUT] | psum | psq | [bf16 planes]
  char*  wb   = (char*)d_ws;
  float* s    = (float*)wb;
  float* t    = s + IN;
  float* cb   = t + IN;
  float* psum = cb + OUTF;
  float* psq  = psum + (size_t)SPLITS * IN;

  const size_t small_bytes  = ((size_t)2 * IN + OUTF + (size_t)2 * SPLITS * IN) * sizeof(float);
  const size_t planes_off   = (small_bytes + 255) & ~(size_t)255;
  const size_t xplane_elems = (size_t)BATCH * IN;
  const size_t wplane_elems = (size_t)OUTF * IN;
  const size_t planes_bytes = 2 * (xplane_elems + wplane_elems) * sizeof(unsigned short);
  const bool   use_packed   = ws_size >= planes_off + planes_bytes;

  bn_partial_kernel <<<dim3(IN / 256, SPLITS), 256, 0, stream>>>(x, psum, psq);
  bn_finalize_kernel<<<IN / 256, 256, 0, stream>>>(psum, psq, gamma, beta, s, t);
  fold_bias_kernel  <<<OUTF, 256, 0, stream>>>(t, W, bias, cb);

  if (use_packed) {
    unsigned short* HA = (unsigned short*)(wb + planes_off);
    unsigned short* LA = HA + xplane_elems;
    unsigned short* HB = LA + xplane_elems;
    unsigned short* LB = HB + wplane_elems;
    pack_x_kernel<<<(unsigned)(xplane_elems / (256 * 4)), 256, 0, stream>>>(x, s, HA, LA);
    pack_w_kernel<<<(unsigned)(wplane_elems / (256 * 4)), 256, 0, stream>>>(W, HB, LB);
    gemm_packed_wmma_kernel<<<dim3(OUTF / BN_, BATCH / BM), 256, 0, stream>>>(
        HA, LA, HB, LB, cb, out);
  } else {
    gemm_bn_wmma_kernel<<<dim3(OUTF / BN_, BATCH / BM), 256, 0, stream>>>(
        x, W, s, cb, out);
  }
}
